// SpatialTemporalModel_27307402068082
// MI455X (gfx1250) — compile-verified, matrix operand order checked by reading, not run on hardware
//
#include <hip/hip_runtime.h>
#include <hip/hip_bf16.h>

// ---------------------------------------------------------------------------
// Types for CDNA5 WMMA (wave32): v_wmma_f32_16x16x32_bf16
// ---------------------------------------------------------------------------
typedef __attribute__((ext_vector_type(16))) __bf16 v16bf;
typedef __attribute__((ext_vector_type(8)))  float  v8f;

union FragBF {
  v16bf v;
  uint4 q[2];
};

// 16-bit A/B fragment layout (ISA 7.12.2): lane l covers M/N = l&15,
// per-lane K elements = base + {0..7} and base + {16..23}, base = (l<16?0:8).
// Both halves are 8 contiguous bf16 -> two 16-byte loads per fragment.
__device__ __forceinline__ v16bf load_frag_bf16(const __bf16* rowPtr, int base) {
  FragBF f;
  f.q[0] = *(const uint4*)(rowPtr + base);
  f.q[1] = *(const uint4*)(rowPtr + base + 16);
  return f.v;
}

// ---------------------------------------------------------------------------
// Utility kernels
// ---------------------------------------------------------------------------
__global__ void zero_u32(unsigned int* __restrict__ p, long long n) {
  long long i = (long long)blockIdx.x * blockDim.x + threadIdx.x;
  if (i < n) p[i] = 0u;
}

__global__ void deg_count(const int* __restrict__ dst, float* __restrict__ deg, int E) {
  int i = blockIdx.x * blockDim.x + threadIdx.x;
  if (i < E)
    __hip_atomic_fetch_add(&deg[dst[i]], 1.0f, __ATOMIC_RELAXED, __HIP_MEMORY_SCOPE_AGENT);
}

__global__ void deg_to_dinv(float* __restrict__ deg, int N) {
  int i = blockIdx.x * blockDim.x + threadIdx.x;
  if (i < N) deg[i] = rsqrtf(deg[i] + 1.0f);
}

// ---------------------------------------------------------------------------
// Weight preparation (f32 -> bf16, transpose / concat / pad once per call)
// ---------------------------------------------------------------------------
// W1 (16x64) -> w1tp (64x32), w1tp[j][k] = W1[k][j], zero-padded K in [16,32)
__global__ void prep_w1(const float* __restrict__ W1, __bf16* __restrict__ w1tp) {
  int tid = blockIdx.x * blockDim.x + threadIdx.x;
  if (tid >= 64 * 32) return;
  int j = tid >> 5, k = tid & 31;
  w1tp[j * 32 + k] = (k < 16) ? (__bf16)W1[k * 64 + j] : (__bf16)0.0f;
}

// W2 (64x64) -> w2t (64x64), w2t[j][k] = W2[k][j]
__global__ void prep_w2(const float* __restrict__ W2, __bf16* __restrict__ w2t) {
  int tid = blockIdx.x * blockDim.x + threadIdx.x;
  if (tid >= 64 * 64) return;
  int j = tid >> 6, k = tid & 63;
  w2t[j * 64 + k] = (__bf16)W2[k * 64 + j];
}

// BT (512x192): row j = [W_ih row j (64) | W_hh row j (128)]; bsum = b_ih + b_hh
__global__ void prep_lstm(const float* __restrict__ Wih, const float* __restrict__ Whh,
                          const float* __restrict__ bih, const float* __restrict__ bhh,
                          __bf16* __restrict__ BT, float* __restrict__ bsum) {
  int tid = blockIdx.x * blockDim.x + threadIdx.x;
  if (tid >= 512 * 192) return;
  int j = tid / 192, k = tid - j * 192;
  BT[j * 192 + k] = (k < 64) ? (__bf16)Wih[j * 64 + k] : (__bf16)Whh[j * 128 + (k - 64)];
  if (k == 0) bsum[j] = bih[j] + bhh[j];
}

// ---------------------------------------------------------------------------
// GCN layer-1 GEMM: out(Nx64 f32) = x_t(Nx16 f32) @ W1, K padded to 32.
// One wave per 16-row strip: A loaded ONCE, 4 column tiles -> 4 WMMAs.
// ---------------------------------------------------------------------------
__global__ void __launch_bounds__(256) gemm_x_w1(const float* __restrict__ x,
                                                 const __bf16* __restrict__ w1tp,
                                                 float* __restrict__ out, int mTiles) {
  const int wave = threadIdx.x >> 5;
  const int lane = threadIdx.x & 31;
  const int mt = blockIdx.x * 8 + wave;
  if (mt >= mTiles) return;                  // wave-uniform: EXEC stays all-ones
  const int m0   = mt << 4;
  const int half = lane >> 4;
  const int base = half << 3;                // per-lane K base: 0 or 8
  const int row  = m0 + (lane & 15);

  // A: 8 contiguous f32 of x row at col base -> bf16; K in [16,32) is zero pad
  const float* xr = x + (size_t)row * 16 + base;
  float4 p0 = *(const float4*)(xr);
  float4 p1 = *(const float4*)(xr + 4);
  v16bf a;
  a[0] = (__bf16)p0.x; a[1] = (__bf16)p0.y; a[2] = (__bf16)p0.z; a[3] = (__bf16)p0.w;
  a[4] = (__bf16)p1.x; a[5] = (__bf16)p1.y; a[6] = (__bf16)p1.z; a[7] = (__bf16)p1.w;
#pragma unroll
  for (int e = 8; e < 16; ++e) a[e] = (__bf16)0.0f;

  v8f c[4];
#pragma unroll
  for (int jt = 0; jt < 4; ++jt) {
#pragma unroll
    for (int r = 0; r < 8; ++r) c[jt][r] = 0.0f;
    v16bf b = load_frag_bf16(w1tp + (size_t)((jt << 4) + (lane & 15)) * 32, base);
    c[jt] = __builtin_amdgcn_wmma_f32_16x16x32_bf16(false, a, false, b, (short)0,
                                                    c[jt], false, false);
  }

#pragma unroll
  for (int jt = 0; jt < 4; ++jt)
#pragma unroll
    for (int r = 0; r < 8; ++r)
      out[(size_t)(m0 + r + (half << 3)) * 64 + (jt << 4) + (lane & 15)] = c[jt][r];
}

// ---------------------------------------------------------------------------
// GCN layer-2 GEMM: out(Nx64 f32) = h1(Nx64 bf16) @ W2, K = 64 (2 chunks).
// One wave per 16-row strip: A shared across 4 column tiles -> 8 WMMAs/wave.
// ---------------------------------------------------------------------------
__global__ void __launch_bounds__(256) gemm_h_w2(const __bf16* __restrict__ h,
                                                 const __bf16* __restrict__ w2t,
                                                 float* __restrict__ out, int mTiles) {
  const int wave = threadIdx.x >> 5;
  const int lane = threadIdx.x & 31;
  const int mt = blockIdx.x * 8 + wave;
  if (mt >= mTiles) return;
  const int m0   = mt << 4;
  const int half = lane >> 4;
  const int base = half << 3;
  const int row  = m0 + (lane & 15);

  v8f c[4];
#pragma unroll
  for (int jt = 0; jt < 4; ++jt)
#pragma unroll
    for (int r = 0; r < 8; ++r) c[jt][r] = 0.0f;

#pragma unroll
  for (int kc = 0; kc < 2; ++kc) {
    const int K0 = kc << 5;
    v16bf a = load_frag_bf16(h + (size_t)row * 64 + K0, base);
#pragma unroll
    for (int jt = 0; jt < 4; ++jt) {
      v16bf b = load_frag_bf16(w2t + (size_t)((jt << 4) + (lane & 15)) * 64 + K0, base);
      c[jt] = __builtin_amdgcn_wmma_f32_16x16x32_bf16(false, a, false, b, (short)0,
                                                      c[jt], false, false);
    }
  }

#pragma unroll
  for (int jt = 0; jt < 4; ++jt)
#pragma unroll
    for (int r = 0; r < 8; ++r)
      out[(size_t)(m0 + r + (half << 3)) * 64 + (jt << 4) + (lane & 15)] = c[jt][r];
}

// ---------------------------------------------------------------------------
// Edge aggregation: out[dst] += xw[src] * dinv[src]*dinv[dst] (64 ch / edge)
// 16 threads/edge, float4 gather + 4 native f32 atomic scatters each.
// ---------------------------------------------------------------------------
__global__ void edge_agg(const float* __restrict__ xw, const float* __restrict__ dinv,
                         const int* __restrict__ src, const int* __restrict__ dst,
                         float* __restrict__ out, int E) {
  long long tid = (long long)blockIdx.x * blockDim.x + threadIdx.x;
  if (tid >= (long long)E * 16) return;
  const int e  = (int)(tid >> 4);
  const int c4 = (int)(tid & 15) << 2;
  const int s = src[e], d = dst[e];
  const float w = dinv[s] * dinv[d];
  const float4 v = *(const float4*)(xw + (size_t)s * 64 + c4);
  float* o = out + (size_t)d * 64 + c4;
  __hip_atomic_fetch_add(o + 0, v.x * w, __ATOMIC_RELAXED, __HIP_MEMORY_SCOPE_AGENT);
  __hip_atomic_fetch_add(o + 1, v.y * w, __ATOMIC_RELAXED, __HIP_MEMORY_SCOPE_AGENT);
  __hip_atomic_fetch_add(o + 2, v.z * w, __ATOMIC_RELAXED, __HIP_MEMORY_SCOPE_AGENT);
  __hip_atomic_fetch_add(o + 3, v.w * w, __ATOMIC_RELAXED, __HIP_MEMORY_SCOPE_AGENT);
}

// h = relu(agg + xw * dinv^2 + bias): 4 channels/thread, float4 in, 4xbf16 out
__global__ void relu_combine(const float* __restrict__ agg, const float* __restrict__ xw,
                             const float* __restrict__ dinv, const float* __restrict__ bias,
                             __bf16* __restrict__ out, int N) {
  long long tid = (long long)blockIdx.x * blockDim.x + threadIdx.x;
  if (tid >= (long long)N * 16) return;
  const int node = (int)(tid >> 4);
  const int c    = (int)(tid & 15) << 2;
  const float di = dinv[node];
  const float d2 = di * di;
  const size_t o = (size_t)node * 64 + c;
  const float4 va = *(const float4*)(agg + o);
  const float4 vx = *(const float4*)(xw + o);
  const float4 vb = *(const float4*)(bias + c);
  union {
    __bf16 h[4];
    uint2  u;
  } pk;
  pk.h[0] = (__bf16)fmaxf(va.x + vx.x * d2 + vb.x, 0.0f);
  pk.h[1] = (__bf16)fmaxf(va.y + vx.y * d2 + vb.y, 0.0f);
  pk.h[2] = (__bf16)fmaxf(va.z + vx.z * d2 + vb.z, 0.0f);
  pk.h[3] = (__bf16)fmaxf(va.w + vx.w * d2 + vb.w, 0.0f);
  *(uint2*)(out + o) = pk.u;
}

// ---------------------------------------------------------------------------
// Fused LSTM step. Block = 8 waves handles 16 nodes x all 512 gate columns:
//   gates = [h_t | h_prev] @ BT^T + bsum  (WMMA bf16, f32 acc, K = 192)
// Gates stay in LDS; sigmoid/tanh cell update done in-place. Gates never hit HBM.
// ---------------------------------------------------------------------------
#define GSTRIDE 520
__global__ void __launch_bounds__(256) lstm_step_kernel(
    const __bf16* __restrict__ ht,    // N x 64   (GCN output, this step)
    __bf16* __restrict__ hprev,       // N x 128  (read as A, then overwritten)
    float* __restrict__ cstate,       // N x 128
    const __bf16* __restrict__ BT,    // 512 x 192
    const float* __restrict__ bsum) { // 512
  __shared__ float gates[16 * GSTRIDE];
  const int wave = threadIdx.x >> 5;
  const int lane = threadIdx.x & 31;
  const int m0   = blockIdx.x << 4;
  const int half = lane >> 4;
  const int base = half << 3;
  const int row  = m0 + (lane & 15);

  v8f acc[4];
#pragma unroll
  for (int tt = 0; tt < 4; ++tt)
#pragma unroll
    for (int r = 0; r < 8; ++r) acc[tt][r] = 0.0f;

#pragma unroll
  for (int kc = 0; kc < 6; ++kc) {
    const int K0 = kc << 5;
    v16bf a = (K0 < 64)
        ? load_frag_bf16(ht    + (size_t)row * 64  + K0,        base)
        : load_frag_bf16(hprev + (size_t)row * 128 + (K0 - 64), base);
#pragma unroll
    for (int tt = 0; tt < 4; ++tt) {
      const int col = (wave << 6) + (tt << 4) + (lane & 15);
      v16bf b = load_frag_bf16(BT + (size_t)col * 192 + K0, base);
      acc[tt] = __builtin_amdgcn_wmma_f32_16x16x32_bf16(false, a, false, b, (short)0,
                                                        acc[tt], false, false);
    }
  }

  // Spill biased gates to LDS (C layout: reg r -> row r + 8*half, col = lane&15)
#pragma unroll
  for (int tt = 0; tt < 4; ++tt) {
    const int col = (wave << 6) + (tt << 4) + (lane & 15);
    const float bb = bsum[col];
#pragma unroll
    for (int r = 0; r < 8; ++r)
      gates[(r + (half << 3)) * GSTRIDE + col] = acc[tt][r] + bb;
  }
  __syncthreads();

  // Elementwise LSTM update (torch gate order i,f,g,o)
  for (int idx = threadIdx.x; idx < 16 * 128; idx += 256) {
    const int rr = idx >> 7;
    const int j  = idx & 127;
    const int node = m0 + rr;
    const float gi = gates[rr * GSTRIDE + j];
    const float gf = gates[rr * GSTRIDE + 128 + j];
    const float gg = gates[rr * GSTRIDE + 256 + j];
    const float go = gates[rr * GSTRIDE + 384 + j];
    const float si = 1.0f / (1.0f + __expf(-gi));
    const float sf = 1.0f / (1.0f + __expf(-gf));
    const float tg = tanhf(gg);
    const float so = 1.0f / (1.0f + __expf(-go));
    const float cn = sf * cstate[(size_t)node * 128 + j] + si * tg;
    cstate[(size_t)node * 128 + j] = cn;
    hprev [(size_t)node * 128 + j] = (__bf16)(so * tanhf(cn));
  }
}

// ---------------------------------------------------------------------------
// Output head: out = relu(hT @ Wf1 + bf1) @ Wf2 + bf2. Weights in LDS.
// ---------------------------------------------------------------------------
__global__ void __launch_bounds__(256) head_mlp(const __bf16* __restrict__ h,
                                                const float* __restrict__ Wf1,
                                                const float* __restrict__ bf1,
                                                const float* __restrict__ Wf2,
                                                const float* __restrict__ bf2,
                                                float* __restrict__ out, int N) {
  __shared__ float w1s[128 * 32];
  __shared__ float b1s[32];
  __shared__ float w2s[32];
  for (int i = threadIdx.x; i < 128 * 32; i += 256) w1s[i] = Wf1[i];
  if (threadIdx.x < 32) {
    b1s[threadIdx.x] = bf1[threadIdx.x];
    w2s[threadIdx.x] = Wf2[threadIdx.x];
  }
  __syncthreads();
  const int n = blockIdx.x * 256 + threadIdx.x;
  if (n >= N) return;

  float acc[32];
#pragma unroll
  for (int j = 0; j < 32; ++j) acc[j] = b1s[j];
  for (int k = 0; k < 128; ++k) {
    const float hv = (float)h[(size_t)n * 128 + k];
#pragma unroll
    for (int j = 0; j < 32; ++j) acc[j] += hv * w1s[k * 32 + j];
  }
  float r = bf2[0];
#pragma unroll
  for (int j = 0; j < 32; ++j) r += fmaxf(acc[j], 0.0f) * w2s[j];
  out[n] = r;
}

// ---------------------------------------------------------------------------
// Host orchestration
// ---------------------------------------------------------------------------
extern "C" void kernel_launch(void* const* d_in, const int* in_sizes, int n_in,
                              void* d_out, int out_size, void* d_ws, size_t ws_size,
                              hipStream_t stream) {
  const float* x_seq = (const float*)d_in[0];
  const int*   eidx  = (const int*)d_in[1];
  const float* W1  = (const float*)d_in[2];
  const float* b1  = (const float*)d_in[3];
  const float* W2  = (const float*)d_in[4];
  const float* b2  = (const float*)d_in[5];
  const float* Wih = (const float*)d_in[6];
  const float* Whh = (const float*)d_in[7];
  const float* bih = (const float*)d_in[8];
  const float* bhh = (const float*)d_in[9];
  const float* Wf1 = (const float*)d_in[10];
  const float* bf1v = (const float*)d_in[11];
  const float* Wf2 = (const float*)d_in[12];
  const float* bf2v = (const float*)d_in[13];
  float* out = (float*)d_out;

  const int T = 24, F = 16, H = 64, LH = 128;
  const int E = in_sizes[1] / 2;
  const int N = in_sizes[0] / (T * F);   // 50000, multiple of 16
  const int* src = eidx;
  const int* dst = eidx + E;

  // Workspace carving (256B aligned): ~77 MB total
  char* ws = (char*)d_ws;
  size_t off = 0;
  auto carve = [&](size_t bytes) -> void* {
    void* p = ws + off;
    off += (bytes + 255) & ~(size_t)255;
    return p;
  };
  float*  dinv   = (float*) carve((size_t)N * 4);
  float*  bufXW  = (float*) carve((size_t)N * H * 4);
  float*  bufAGG = (float*) carve((size_t)N * H * 4);
  __bf16* h1bf   = (__bf16*)carve((size_t)N * H * 2);
  __bf16* htbf   = (__bf16*)carve((size_t)N * H * 2);
  __bf16* hprev  = (__bf16*)carve((size_t)N * LH * 2);
  float*  cst    = (float*) carve((size_t)N * LH * 4);
  __bf16* w1tp   = (__bf16*)carve((size_t)H * 32 * 2);
  __bf16* w2t    = (__bf16*)carve((size_t)H * H * 2);
  __bf16* BT     = (__bf16*)carve((size_t)(4 * LH) * (H + LH) * 2);
  float*  bsum   = (float*) carve((size_t)(4 * LH) * 4);

  auto cdiv = [](long long a, long long b) { return (unsigned)((a + b - 1) / b); };

  // Degree -> D^{-1/2}
  zero_u32<<<cdiv(N, 256), 256, 0, stream>>>((unsigned*)dinv, (long long)N);
  deg_count<<<cdiv(E, 256), 256, 0, stream>>>(dst, dinv, E);
  deg_to_dinv<<<cdiv(N, 256), 256, 0, stream>>>(dinv, N);

  // Weight prep (bf16 transpose/concat/pad)
  prep_w1<<<cdiv(64 * 32, 256), 256, 0, stream>>>(W1, w1tp);
  prep_w2<<<cdiv(64 * 64, 256), 256, 0, stream>>>(W2, w2t);
  prep_lstm<<<cdiv(512 * 192, 256), 256, 0, stream>>>(Wih, Whh, bih, bhh, BT, bsum);

  // LSTM state init
  zero_u32<<<cdiv((long long)N * LH, 256), 256, 0, stream>>>((unsigned*)cst, (long long)N * LH);
  zero_u32<<<cdiv((long long)N * LH / 2, 256), 256, 0, stream>>>((unsigned*)hprev,
                                                                 (long long)N * LH / 2);

  const int mTiles = N / 16;
  const unsigned gemmBlocks = cdiv(mTiles, 8);
  const unsigned ewBlocks   = cdiv((long long)N * 16, 256);
  const unsigned aggBlocks  = cdiv((long long)E * 16, 256);
  const unsigned zeroBlocks = cdiv((long long)N * H, 256);

  for (int t = 0; t < T; ++t) {
    const float* xt = x_seq + (size_t)t * N * F;
    // GCN layer 1
    gemm_x_w1<<<gemmBlocks, 256, 0, stream>>>(xt, w1tp, bufXW, mTiles);
    zero_u32<<<zeroBlocks, 256, 0, stream>>>((unsigned*)bufAGG, (long long)N * H);
    edge_agg<<<aggBlocks, 256, 0, stream>>>(bufXW, dinv, src, dst, bufAGG, E);
    relu_combine<<<ewBlocks, 256, 0, stream>>>(bufAGG, bufXW, dinv, b1, h1bf, N);
    // GCN layer 2
    gemm_h_w2<<<gemmBlocks, 256, 0, stream>>>(h1bf, w2t, bufXW, mTiles);
    zero_u32<<<zeroBlocks, 256, 0, stream>>>((unsigned*)bufAGG, (long long)N * H);
    edge_agg<<<aggBlocks, 256, 0, stream>>>(bufXW, dinv, src, dst, bufAGG, E);
    relu_combine<<<ewBlocks, 256, 0, stream>>>(bufAGG, bufXW, dinv, b2, htbf, N);
    // Fused LSTM step (gates in LDS, state updated in-place)
    lstm_step_kernel<<<mTiles, 256, 0, stream>>>(htbf, hprev, cst, BT, bsum);
  }

  head_mlp<<<cdiv(N, 256), 256, 0, stream>>>(hprev, Wf1, bf1v, Wf2, bf2v, out, N);
}
